// ChannelMaskedAttention_31756988186887
// MI455X (gfx1250) — matrix operand order repaired and checked by reading, not compiled, not run on hardware
//
#include <hip/hip_runtime.h>

typedef _Float16 half8   __attribute__((ext_vector_type(8)));
typedef _Float16 half16  __attribute__((ext_vector_type(16)));
typedef float    floatx8 __attribute__((ext_vector_type(8)));
typedef unsigned int uint32x4 __attribute__((ext_vector_type(4)));
typedef int          int32x4  __attribute__((ext_vector_type(4)));
typedef int          int32x8  __attribute__((ext_vector_type(8)));

#define DIMX 512
#define NSEQ 64
#define DHEAD 64
#define NBATCH 2048

// LDS layout (halves): xa 0, qs 32768, ks 65536, vt 98304, wt[2] 131072 (2 x 16384)
#define WT_HALF_OFF   131072
#define WT_BYTE_OFF   262144u
#define WT_BYTE_SZ    32768u
#define LDS_TOTAL_B   327680   // 320 KB

#if __has_builtin(__builtin_amdgcn_tensor_load_to_lds)
#define USE_TDM 1
#else
#define USE_TDM 0
#endif

__device__ __forceinline__ half16 ldfrag(const _Float16* p0, const _Float16* p1) {
    union { half16 v; half8 h[2]; } u;
    u.h[0] = *(const half8*)p0;
    u.h[1] = *(const half8*)p1;
    return u.v;
}

__device__ __forceinline__ floatx8 wmma_f16(half16 a, half16 b, floatx8 c) {
    return __builtin_amdgcn_wmma_f32_16x16x32_f16(false, a, false, b, (short)0, c, false, false);
}

#define SWZ(x, m) __uint_as_float((unsigned)__builtin_amdgcn_ds_swizzle( \
        (int)__float_as_uint(x), (((m) << 10) | 0x1f)))

#if USE_TDM
// 2-D TDM tile load: 512 rows x 32 cols of f16 (n-major weights), packed into LDS tile buffer.
__device__ __forceinline__ void tdm_load_tile(const _Float16* gtile, unsigned lds_byte) {
    unsigned long long ga = (unsigned long long)(uintptr_t)gtile;
    uint32x4 g0 = { 1u,                                   // count=1, user mode
                    lds_byte,                             // lds_addr (bytes)
                    (unsigned)ga,                         // global_addr[31:0]
                    (unsigned)((ga >> 32) & 0x1FFFFFFu) | (2u << 30) }; // addr[56:32] | type=2
    int32x8 g1 = { (int)(1u << 16),     // data_size = 1 (2 bytes)
                   (int)(512u << 16),   // tensor_dim0 = 512 (low16 in [63:48])
                   (int)(512u << 16),   // tensor_dim0 hi=0 ; tensor_dim1 = 512 (low16)
                   (int)(32u << 16),    // tensor_dim1 hi=0 ; tile_dim0 = 32
                   (int)512,            // tile_dim1 = 512 ; tile_dim2 = 0
                   (int)512,            // tensor_dim0_stride = 512
                   0, 0 };
    int32x4 z4 = { 0, 0, 0, 0 };
    int32x8 z8 = { 0, 0, 0, 0, 0, 0, 0, 0 };
    // clang-23 / therock-10.0 lane: 6-arg form (g0, g1, g2, g3, extra, cpol)
    __builtin_amdgcn_tensor_load_to_lds(g0, g1, z4, z4, z8, 0);
}
#endif

// Pre-convert one 512x512 f32 weight (k-major) to f16 transposed (n-major).
__global__ __launch_bounds__(256)
void wconv_kernel(const float* __restrict__ W, _Float16* __restrict__ dst) {
    const int n = blockIdx.x;
    for (int k = threadIdx.x; k < DIMX; k += 256)
        dst[n * DIMX + k] = (_Float16)W[k * DIMX + n];
}

__global__ __launch_bounds__(256)
void cma_fused_kernel(const float* __restrict__ xg,
                      const float* __restrict__ cmask,
                      const _Float16* __restrict__ wsT,   // 4 x [512][512] f16, n-major
                      const float* __restrict__ bq, const float* __restrict__ bk,
                      const float* __restrict__ bv, const float* __restrict__ bo,
                      float* __restrict__ out)
{
    extern __shared__ _Float16 smem[];
    _Float16* xa = smem;                  // [64][512] x; reused as attn-out
    _Float16* qs = smem + 32768;          // [64][512]
    _Float16* ks = smem + 65536;          // [64][512]
    _Float16* vt = smem + 98304;          // [512][64] v transposed
    _Float16* wt = smem + WT_HALF_OFF;    // 2 x [512][32] tile buffers; P scratch in phase B

    const int tid  = threadIdx.x;
    const int wv   = tid >> 5;
    const int lane = tid & 31;
    const int ln   = lane & 15;
    const int hi   = lane >> 4;
    const int b    = blockIdx.x;

    const float* xb = xg + (size_t)b * (NSEQ * DIMX);
    for (int idx = tid; idx < NSEQ * DIMX; idx += 256)
        xa[idx] = (_Float16)xb[idx];
    __syncthreads();

    const int mt = wv & 3;
    const int nh = wv >> 2;
    const floatx8 vzero = {};

    // -------- phase A: q/k/v projections (M=64,N=512,K=512), TDM-streamed W tiles --------
#pragma unroll 1
    for (int p = 0; p < 3; ++p) {
        const _Float16* wsrc = wsT + (size_t)p * (DIMX * DIMX);
        const float* Bi = (p == 0) ? bq : (p == 1) ? bk : bv;
        floatx8 acc[16];
#pragma unroll
        for (int nt = 0; nt < 16; ++nt) acc[nt] = vzero;

#if USE_TDM
        if (wv == 0) tdm_load_tile(wsrc, WT_BYTE_OFF);   // kb=0 -> buf0
#endif
#pragma unroll 1
        for (int kb = 0; kb < 16; ++kb) {
            const int buf = kb & 1;
#if USE_TDM
            if (wv == 0) __builtin_amdgcn_s_wait_tensorcnt(0);
            __syncthreads();
            if (wv == 0 && kb < 15)
                tdm_load_tile(wsrc + (kb + 1) * 32, WT_BYTE_OFF + (buf ^ 1) * WT_BYTE_SZ);
#else
            __syncthreads();
            for (int idx = tid * 8; idx < 512 * 32; idx += 256 * 8)
                *(half8*)(wt + buf * 16384 + idx) =
                    *(const half8*)(wsrc + ((idx >> 5) * DIMX) + kb * 32 + (idx & 31));
            __syncthreads();
#endif
            const _Float16* ap = xa + (mt * 16 + ln) * DIMX + kb * 32 + 16 * hi;
            half16 a = ldfrag(ap, ap + 8);
            const _Float16* bbase = wt + buf * 16384 + (nh * 256 + ln) * 32 + 16 * hi;
            half16 bf[2];
            bf[0] = ldfrag(bbase, bbase + 8);
#pragma unroll
            for (int nt = 0; nt < 16; ++nt) {
                if (nt < 15) {
                    const _Float16* bp = bbase + (nt + 1) * 512;
                    bf[(nt + 1) & 1] = ldfrag(bp, bp + 8);
                }
                acc[nt] = wmma_f16(a, bf[nt & 1], acc[nt]);
            }
#if USE_TDM
            __syncthreads();   // tile consumed before next wait/issue reuses it
#endif
        }
        _Float16* dst = (p == 0) ? qs : (p == 1) ? ks : vt;
#pragma unroll
        for (int nt = 0; nt < 16; ++nt) {
            int ncol = nh * 256 + nt * 16 + ln;
            float bias = Bi[ncol];
#pragma unroll
            for (int r = 0; r < 8; ++r) {
                int row = mt * 16 + r + 8 * hi;
                float val = acc[nt][r] + bias;
                if (p < 2) dst[row * DIMX + ncol] = (_Float16)val;
                else       dst[ncol * NSEQ + row] = (_Float16)val;
            }
        }
    }
    __syncthreads();

    // -------- phase B: attention, wave w owns head h = w --------
    const int h = wv;
    const float* cm = cmask + (size_t)b * (NSEQ * NSEQ);
    _Float16* pl = wt + wv * 1024;   // per-wave 16x64 P scratch (buf0 region, free now)

#pragma unroll 1
    for (int m = 0; m < 4; ++m) {
        floatx8 s[4];
#pragma unroll
        for (int nt = 0; nt < 4; ++nt) s[nt] = vzero;
#pragma unroll
        for (int kk = 0; kk < 2; ++kk) {
            const _Float16* ap = qs + (m * 16 + ln) * DIMX + h * DHEAD + kk * 32 + 16 * hi;
            half16 a = ldfrag(ap, ap + 8);
#pragma unroll
            for (int nt = 0; nt < 4; ++nt) {
                const _Float16* bp = ks + (nt * 16 + ln) * DIMX + h * DHEAD + kk * 32 + 16 * hi;
                s[nt] = wmma_f16(a, ldfrag(bp, bp + 8), s[nt]);
            }
        }
        float mx[8], sum[8];
#pragma unroll
        for (int r = 0; r < 8; ++r) mx[r] = -3.0e38f;
#pragma unroll
        for (int nt = 0; nt < 4; ++nt) {
            int j = nt * 16 + ln;
#pragma unroll
            for (int r = 0; r < 8; ++r) {
                int i = m * 16 + r + 8 * hi;
                float v = s[nt][r] * 0.125f + __logf(cm[i * NSEQ + j] + 1e-9f);
                s[nt][r] = v;
                mx[r] = fmaxf(mx[r], v);
            }
        }
#pragma unroll
        for (int r = 0; r < 8; ++r) {
            mx[r] = fmaxf(mx[r], SWZ(mx[r], 1));
            mx[r] = fmaxf(mx[r], SWZ(mx[r], 2));
            mx[r] = fmaxf(mx[r], SWZ(mx[r], 4));
            mx[r] = fmaxf(mx[r], SWZ(mx[r], 8));
            sum[r] = 0.f;
        }
#pragma unroll
        for (int nt = 0; nt < 4; ++nt)
#pragma unroll
            for (int r = 0; r < 8; ++r) {
                float e = __expf(s[nt][r] - mx[r]);
                s[nt][r] = e;
                sum[r] += e;
            }
#pragma unroll
        for (int r = 0; r < 8; ++r) {
            sum[r] += SWZ(sum[r], 1);
            sum[r] += SWZ(sum[r], 2);
            sum[r] += SWZ(sum[r], 4);
            sum[r] += SWZ(sum[r], 8);
            sum[r] = 1.0f / sum[r];
        }
#pragma unroll
        for (int nt = 0; nt < 4; ++nt)
#pragma unroll
            for (int r = 0; r < 8; ++r)
                pl[(r + 8 * hi) * 64 + nt * 16 + ln] = (_Float16)(s[nt][r] * sum[r]);
        floatx8 o[4];
#pragma unroll
        for (int nt = 0; nt < 4; ++nt) o[nt] = vzero;
#pragma unroll
        for (int kk = 0; kk < 2; ++kk) {
            const _Float16* ap = pl + ln * 64 + kk * 32 + 16 * hi;
            half16 a = ldfrag(ap, ap + 8);
#pragma unroll
            for (int nt = 0; nt < 4; ++nt) {
                const _Float16* bp = vt + (h * DHEAD + nt * 16 + ln) * NSEQ + kk * 32 + 16 * hi;
                o[nt] = wmma_f16(a, ldfrag(bp, bp + 8), o[nt]);
            }
        }
#pragma unroll
        for (int nt = 0; nt < 4; ++nt)
#pragma unroll
            for (int r = 0; r < 8; ++r)
                xa[(m * 16 + r + 8 * hi) * DIMX + h * DHEAD + nt * 16 + ln] = (_Float16)o[nt][r];
    }
    __syncthreads();

    // -------- phase C: output projection O = ao @ Wo + bo --------
    {
        const _Float16* wsrc = wsT + (size_t)3 * (DIMX * DIMX);
        floatx8 acc[16];
#pragma unroll
        for (int nt = 0; nt < 16; ++nt) acc[nt] = vzero;
#if USE_TDM
        if (wv == 0) tdm_load_tile(wsrc, WT_BYTE_OFF);
#endif
#pragma unroll 1
        for (int kb = 0; kb < 16; ++kb) {
            const int buf = kb & 1;
#if USE_TDM
            if (wv == 0) __builtin_amdgcn_s_wait_tensorcnt(0);
            __syncthreads();
            if (wv == 0 && kb < 15)
                tdm_load_tile(wsrc + (kb + 1) * 32, WT_BYTE_OFF + (buf ^ 1) * WT_BYTE_SZ);
#else
            __syncthreads();
            for (int idx = tid * 8; idx < 512 * 32; idx += 256 * 8)
                *(half8*)(wt + buf * 16384 + idx) =
                    *(const half8*)(wsrc + ((idx >> 5) * DIMX) + kb * 32 + (idx & 31));
            __syncthreads();
#endif
            const _Float16* ap = xa + (mt * 16 + ln) * DIMX + kb * 32 + 16 * hi;
            half16 a = ldfrag(ap, ap + 8);
            const _Float16* bbase = wt + buf * 16384 + (nh * 256 + ln) * 32 + 16 * hi;
            half16 bf[2];
            bf[0] = ldfrag(bbase, bbase + 8);
#pragma unroll
            for (int nt = 0; nt < 16; ++nt) {
                if (nt < 15) {
                    const _Float16* bp = bbase + (nt + 1) * 512;
                    bf[(nt + 1) & 1] = ldfrag(bp, bp + 8);
                }
                acc[nt] = wmma_f16(a, bf[nt & 1], acc[nt]);
            }
#if USE_TDM
            __syncthreads();
#endif
        }
        float* ob = out + (size_t)b * (NSEQ * DIMX);
#pragma unroll
        for (int nt = 0; nt < 16; ++nt) {
            int ncol = nh * 256 + nt * 16 + ln;
            float bias = bo[ncol];
#pragma unroll
            for (int r = 0; r < 8; ++r)
                ob[(mt * 16 + r + 8 * hi) * DIMX + ncol] = acc[nt][r] + bias;
        }
    }

    if (b == 0 && tid == 0)
        out[(size_t)NBATCH * NSEQ * DIMX] = 0.0f;   // ccd_loss
}

extern "C" void kernel_launch(void* const* d_in, const int* in_sizes, int n_in,
                              void* d_out, int out_size, void* d_ws, size_t ws_size,
                              hipStream_t stream) {
    const float* x  = (const float*)d_in[0];
    const float* cm = (const float*)d_in[1];
    const float* Wq = (const float*)d_in[2];
    const float* bq = (const float*)d_in[3];
    const float* Wk = (const float*)d_in[4];
    const float* bk = (const float*)d_in[5];
    const float* Wv = (const float*)d_in[6];
    const float* bv = (const float*)d_in[7];
    const float* Wo = (const float*)d_in[8];
    const float* bo = (const float*)d_in[9];
    float* out = (float*)d_out;

    _Float16* wsT = (_Float16*)d_ws;   // 4 x 512x512 f16 (2 MB), n-major
    wconv_kernel<<<dim3(DIMX), dim3(256), 0, stream>>>(Wq, wsT + 0 * DIMX * DIMX);
    wconv_kernel<<<dim3(DIMX), dim3(256), 0, stream>>>(Wk, wsT + 1 * DIMX * DIMX);
    wconv_kernel<<<dim3(DIMX), dim3(256), 0, stream>>>(Wv, wsT + 2 * DIMX * DIMX);
    wconv_kernel<<<dim3(DIMX), dim3(256), 0, stream>>>(Wo, wsT + 3 * DIMX * DIMX);

    cma_fused_kernel<<<dim3(NBATCH), dim3(256), LDS_TOTAL_B, stream>>>(
        x, cm, wsT, bq, bk, bv, bo, out);
}
